// SpikingSelfAttention_13786845020317
// MI455X (gfx1250) — matrix operand order repaired and checked by reading, not compiled, not run on hardware
//
#include <hip/hip_runtime.h>

// ---------------------------------------------------------------------------
// SpikingSelfAttention for MI455X (gfx1250, wave32, WMMA + async-to-LDS).
// T=4, N=2, C=256, H=W=32 (HW=1024), heads=8, hd=32.
// All GEMMs in bf16 via v_wmma_f32_16x16x32_bf16; the 256MB scores tensor
// never hits HBM: scores -> LIF(over T) -> spikes @ V fused per tile.
// Tile staging uses GLOBAL_LOAD_ASYNC_TO_LDS (ASYNCcnt) with double buffering
// so the DMA for chunk kc+1 overlaps the WMMA stream of chunk kc.
// ---------------------------------------------------------------------------

#define T_   4
#define NB_  2
#define C_   256
#define HW_  1024
#define HD_  32
#define NH_  8
#define TN_  8   // T_*NB_

typedef __attribute__((ext_vector_type(16))) __bf16 v16bf;
typedef __attribute__((ext_vector_type(8)))  float  v8f;

union V16 { v16bf v; uint4 u[2]; };

__device__ inline unsigned short f2bf(float f) {
    union { float f; unsigned u; } x;
    x.f = f;
    unsigned r = x.u + 0x7FFFu + ((x.u >> 16) & 1u);
    return (unsigned short)(r >> 16);
}

// --- CDNA5 async memory->LDS (per-lane, ASYNCcnt tracked, no VGPR data) ----
// Generic pointers to __shared__ carry the LDS byte offset in the low 32 bits.
__device__ inline void async_ld_b128(void* lds, const void* g) {
    asm volatile("global_load_async_to_lds_b128 %0, %1, off"
                 :: "v"((unsigned)(size_t)lds), "v"(g) : "memory");
}
__device__ inline void async_ld_b64(void* lds, const void* g) {
    asm volatile("global_load_async_to_lds_b64 %0, %1, off"
                 :: "v"((unsigned)(size_t)lds), "v"(g) : "memory");
}
__device__ inline void wait_async0() {
    asm volatile("s_wait_asynccnt 0x0" ::: "memory");
}
__device__ inline void lds_wave_fence() {  // wave-local LDS publish
    asm volatile("s_wait_dscnt 0x0" ::: "memory");
    __builtin_amdgcn_wave_barrier();
}

// A operand (16-bit 16x32): lane: M=lane&15, Kbase=half*16; 16 consecutive K.
__device__ inline v16bf load_a16(const unsigned short* rowbase, int half) {
    V16 a;
    a.u[0] = *(const uint4*)(rowbase + half * 16);
    a.u[1] = *(const uint4*)(rowbase + half * 16 + 8);
    return a.v;
}
// B operand (16-bit 32x16): lane: N=lane&15, Kbase=half*16; 16 consecutive K.
__device__ inline v16bf load_b16(const unsigned short* base) {
    V16 b;
    b.u[0] = *(const uint4*)(base);
    b.u[1] = *(const uint4*)(base + 8);
    return b.v;
}

// ------------------------- K0: converts / init -----------------------------

__global__ void k_cvt_x(const float* __restrict__ x, unsigned short* __restrict__ xT) {
    size_t idx = (size_t)blockIdx.x * 256 + threadIdx.x;
    if (idx >= (size_t)TN_ * C_ * HW_) return;
    int pix = (int)(idx % HW_);
    int c   = (int)((idx / HW_) % C_);
    int tn  = (int)(idx / ((size_t)HW_ * C_));
    xT[((size_t)tn * HW_ + pix) * C_ + c] = f2bf(x[idx]);   // [tn][pix][c]
}

__global__ void k_cvt_w(const float* __restrict__ wq, const float* __restrict__ wk,
                        const float* __restrict__ wv, const float* __restrict__ wo,
                        unsigned short* __restrict__ wb) {
    int idx = blockIdx.x * 256 + threadIdx.x;        // 4 * 65536
    int which = idx >> 16, e = idx & 0xFFFF;
    const float* s = (which == 0) ? wq : (which == 1) ? wk : (which == 2) ? wv : wo;
    wb[idx] = f2bf(s[e]);
}

__global__ void k_zero_stats(float* sum, float* sqs) {
    int i = threadIdx.x;
    if (i < C_) { sum[i] = 0.f; sqs[i] = 0.f; }
}

// ------------------- K1: QKV 1x1-conv GEMM (256x1024x256) ------------------
// block=128 (4 waves); grid = (64 pixtiles, 8 tn, 3 {q,k,v}).
__global__ __launch_bounds__(128) void k_qkv(
        const unsigned short* __restrict__ xT,
        const unsigned short* __restrict__ wq, const unsigned short* __restrict__ wk,
        const unsigned short* __restrict__ wv,
        unsigned short* __restrict__ qo, unsigned short* __restrict__ ko,
        unsigned short* __restrict__ vo) {
    __shared__ alignas(16) unsigned short Bt[2][16 * 32];
    const int tid = threadIdx.x, wave = tid >> 5, lane = tid & 31;
    const int half = lane >> 4, col = lane & 15;
    const int pixbase = blockIdx.x * 16;
    const int tn = blockIdx.y, wsel = blockIdx.z;
    const unsigned short* W = (wsel == 0) ? wq : (wsel == 1) ? wk : wv;
    const unsigned short* xrow = xT + ((size_t)tn * HW_ + pixbase) * C_;
    const int st_row = tid >> 3, st_off = (tid & 7) * 4;  // 16 rows x 8B

    auto stage = [&](int cc, int buf) {
        async_ld_b64(&Bt[buf][st_row * 32 + st_off],
                     xrow + (size_t)st_row * C_ + cc * 32 + st_off);
    };
    stage(0, 0);

    v8f acc[4] = {};
    for (int cc = 0; cc < 8; ++cc) {
        wait_async0();
        __syncthreads();
        if (cc + 1 < 8) stage(cc + 1, (cc + 1) & 1);   // overlap with WMMAs
        const int cbase = cc * 32;
        v16bf b = load_b16(&Bt[cc & 1][col * 32 + half * 16]);
        v16bf a[4];
#pragma unroll
        for (int ot = 0; ot < 4; ++ot)
            a[ot] = load_a16(W + (size_t)((wave * 4 + ot) * 16 + col) * C_ + cbase, half);
#pragma unroll
        for (int ot = 0; ot < 4; ++ot)
            acc[ot] = __builtin_amdgcn_wmma_f32_16x16x32_bf16(
                false, a[ot], false, b, (short)0, acc[ot], false, false);
    }
    unsigned short* dst = (wsel == 0) ? qo : (wsel == 1) ? ko : vo;
#pragma unroll
    for (int ot = 0; ot < 4; ++ot) {
#pragma unroll
        for (int r = 0; r < 8; ++r) {
            int o = (wave * 4 + ot) * 16 + r + half * 8;
            int h = o >> 5, d = o & 31;
            int pix = pixbase + col;
            unsigned short val = f2bf(acc[ot][r]);
            size_t addr;
            if (wsel == 2)  // v: [tn][h][hd][pix]
                addr = (((size_t)tn * NH_ + h) * HD_ + d) * HW_ + pix;
            else            // q,k: [tn][h][pix][hd]
                addr = (((size_t)tn * NH_ + h) * HW_ + pix) * HD_ + d;
            dst[addr] = val;
        }
    }
}

// ---------------- K2: fused scores -> LIF(T) -> spikes @ V -----------------
// block=128 (4 waves, one 16-row q tile each); grid = (16 q-supertiles, n*h).
__global__ __launch_bounds__(128) void k_attn(
        const unsigned short* __restrict__ qb, const unsigned short* __restrict__ kb,
        const unsigned short* __restrict__ vb, unsigned short* __restrict__ attn) {
    __shared__ alignas(16) unsigned short Kt[2][T_ * 32 * 32];  // [t][kpos][hd]
    __shared__ alignas(16) unsigned short Vt[2][T_ * 32 * 32];  // [t][hd][kpos]
    __shared__ alignas(16) unsigned short Sp[4][T_ * 16 * 32];  // per wave [t][q][k]
    const int tid = threadIdx.x, wave = tid >> 5, lane = tid & 31;
    const int half = lane >> 4, col = lane & 15;
    const int nh = blockIdx.y, n = nh >> 3, h = nh & 7;
    const int qbase = blockIdx.x * 64 + wave * 16;
    const int st_t = (tid >> 5) & 3, st_row = tid & 31;  // one k-row + one v-row

    // q A operands (constant over k chunks)
    v16bf Aq[T_];
#pragma unroll
    for (int t = 0; t < T_; ++t) {
        const unsigned short* qr =
            qb + (((((size_t)t * NB_ + n) * NH_) + h) * HW_ + qbase + col) * HD_;
        Aq[t] = load_a16(qr, half);
    }

    auto stage = [&](int kc, int buf) {  // 4+4 async b128 per thread (64B rows)
        const int kposb = kc * 32;
        const unsigned short* ksrc =
            kb + (((((size_t)st_t * NB_ + n) * NH_) + h) * HW_ + kposb + st_row) * HD_;
        const unsigned short* vsrc =
            vb + (((((size_t)st_t * NB_ + n) * NH_) + h) * HD_ + st_row) * HW_ + kposb;
        unsigned short* kdst = &Kt[buf][(st_t * 32 + st_row) * 32];
        unsigned short* vdst = &Vt[buf][(st_t * 32 + st_row) * 32];
#pragma unroll
        for (int j = 0; j < 4; ++j) {
            async_ld_b128(kdst + j * 8, ksrc + j * 8);
            async_ld_b128(vdst + j * 8, vsrc + j * 8);
        }
    };
    stage(0, 0);

    v8f oacc[T_][2] = {};
    const float rs = 0.17677669529663687f;  // 1/sqrt(hd)

    for (int kc = 0; kc < 32; ++kc) {
        wait_async0();          // my chunk-kc rows landed
        __syncthreads();        // everyone's landed; prev compute done
        if (kc + 1 < 32) stage(kc + 1, (kc + 1) & 1);  // DMA overlaps WMMAs
        const int buf = kc & 1;

        // scores, all 4 time steps, two 16-wide k subtiles
        v8f sc[T_][2];
#pragma unroll
        for (int t = 0; t < T_; ++t)
#pragma unroll
            for (int sub = 0; sub < 2; ++sub) {
                v16bf b = load_b16(&Kt[buf][(t * 32 + sub * 16 + col) * 32 + half * 16]);
                v8f z = {};
                sc[t][sub] = __builtin_amdgcn_wmma_f32_16x16x32_bf16(
                    false, Aq[t], false, b, (short)0, z, false, false);
            }

        // LIF recurrence over T per element; write spikes (bf16) to LDS
#pragma unroll
        for (int sub = 0; sub < 2; ++sub)
#pragma unroll
            for (int r = 0; r < 8; ++r) {
                float v = 0.f;
#pragma unroll
                for (int t = 0; t < T_; ++t) {
                    float hm = 0.5f * (v + sc[t][sub][r] * rs);  // charge, tau=2
                    float s = (hm >= 0.5f) ? 1.f : 0.f;          // fire
                    v = (s > 0.f) ? 0.f : hm;                    // hard reset
                    Sp[wave][(t * 16 + r + half * 8) * 32 + sub * 16 + col] = f2bf(s);
                }
            }
        lds_wave_fence();  // Sp is wave-private: no block barrier needed

        // out_t += spikes_t @ v_t  (hd split into two 16-col halves)
#pragma unroll
        for (int t = 0; t < T_; ++t) {
            const unsigned short* sp = &Sp[wave][(t * 16 + col) * 32];
            v16bf a = load_a16(sp, half);
#pragma unroll
            for (int hh = 0; hh < 2; ++hh) {
                v16bf b = load_b16(&Vt[buf][(t * 32 + hh * 16 + col) * 32 + half * 16]);
                oacc[t][hh] = __builtin_amdgcn_wmma_f32_16x16x32_bf16(
                    false, a, false, b, (short)0, oacc[t][hh], false, false);
            }
        }
    }
    // store attn transposed: [t][n][pix][C] bf16 (feeds Wo GEMM B staging)
#pragma unroll
    for (int t = 0; t < T_; ++t)
#pragma unroll
        for (int hh = 0; hh < 2; ++hh)
#pragma unroll
            for (int r = 0; r < 8; ++r)
                attn[(((size_t)t * NB_ + n) * HW_ + qbase + r + half * 8) * C_ +
                     h * HD_ + hh * 16 + col] = f2bf(oacc[t][hh][r]);
}

// -------------- K3: Wo GEMM + BatchNorm statistics (sum/sumsq) -------------
__global__ __launch_bounds__(128) void k_proj(
        const unsigned short* __restrict__ attnT, const unsigned short* __restrict__ wo,
        float* __restrict__ proj, float* __restrict__ sum, float* __restrict__ sqs) {
    __shared__ alignas(16) unsigned short Bt[2][16 * 32];
    const int tid = threadIdx.x, wave = tid >> 5, lane = tid & 31;
    const int half = lane >> 4, col = lane & 15;
    const int pixbase = blockIdx.x * 16, tn = blockIdx.y;
    const unsigned short* arow = attnT + ((size_t)tn * HW_ + pixbase) * C_;
    const int st_row = tid >> 3, st_off = (tid & 7) * 4;

    auto stage = [&](int cc, int buf) {
        async_ld_b64(&Bt[buf][st_row * 32 + st_off],
                     arow + (size_t)st_row * C_ + cc * 32 + st_off);
    };
    stage(0, 0);

    v8f acc[4] = {};
    for (int cc = 0; cc < 8; ++cc) {
        wait_async0();
        __syncthreads();
        if (cc + 1 < 8) stage(cc + 1, (cc + 1) & 1);
        const int cbase = cc * 32;
        v16bf b = load_b16(&Bt[cc & 1][col * 32 + half * 16]);
        v16bf a[4];
#pragma unroll
        for (int ot = 0; ot < 4; ++ot)
            a[ot] = load_a16(wo + (size_t)((wave * 4 + ot) * 16 + col) * C_ + cbase, half);
#pragma unroll
        for (int ot = 0; ot < 4; ++ot)
            acc[ot] = __builtin_amdgcn_wmma_f32_16x16x32_bf16(
                false, a[ot], false, b, (short)0, acc[ot], false, false);
    }
#pragma unroll
    for (int ot = 0; ot < 4; ++ot)
#pragma unroll
        for (int r = 0; r < 8; ++r) {
            int o = (wave * 4 + ot) * 16 + r + half * 8;
            int pix = pixbase + col;
            float val = acc[ot][r];
            proj[((size_t)tn * C_ + o) * HW_ + pix] = val;
            float s1 = val, s2 = val * val;
#pragma unroll
            for (int off = 8; off > 0; off >>= 1) {  // lanes 0-15 / 16-31 share o
                s1 += __shfl_xor(s1, off, 16);
                s2 += __shfl_xor(s2, off, 16);
            }
            if (col == 0) { atomicAdd(&sum[o], s1); atomicAdd(&sqs[o], s2); }
        }
}

// ---------------- K4: BatchNorm + LIF over T + residual add ----------------
__global__ void k_bnlif(const float* __restrict__ proj, const float* __restrict__ x,
                        const float* __restrict__ gamma, const float* __restrict__ beta,
                        const float* __restrict__ sum, const float* __restrict__ sqs,
                        float* __restrict__ out) {
    int idx = blockIdx.x * blockDim.x + threadIdx.x;  // over n*C*HW
    if (idx >= NB_ * C_ * HW_) return;
    int pix = idx % HW_;
    int c   = (idx / HW_) % C_;
    int n   = idx / (HW_ * C_);
    const float cnt = (float)(T_ * NB_ * HW_);
    float m = sum[c] / cnt;
    float var = sqs[c] / cnt - m * m;
    float inv = rsqrtf(var + 1e-5f);
    float g = gamma[c] * inv, b = beta[c];
    float v = 0.f;
#pragma unroll
    for (int t = 0; t < T_; ++t) {
        size_t a = (((size_t)t * NB_ + n) * C_ + c) * HW_ + pix;
        float y = g * (proj[a] - m) + b;
        float hm = 0.5f * (v + y);
        float s = (hm >= 1.0f) ? 1.f : 0.f;  // v_threshold = 1.0
        v = (s > 0.f) ? 0.f : hm;
        out[a] = s + x[a];
    }
}

// ---------------------------------------------------------------------------

extern "C" void kernel_launch(void* const* d_in, const int* in_sizes, int n_in,
                              void* d_out, int out_size, void* d_ws, size_t ws_size,
                              hipStream_t stream) {
    const float* x     = (const float*)d_in[0];
    const float* Wq    = (const float*)d_in[1];
    const float* Wk    = (const float*)d_in[2];
    const float* Wv    = (const float*)d_in[3];
    const float* Wo    = (const float*)d_in[4];
    const float* gamma = (const float*)d_in[5];
    const float* beta  = (const float*)d_in[6];
    float* out = (float*)d_out;

    char* ws = (char*)d_ws;
    size_t off = 0;
    auto take = [&](size_t bytes) { char* p = ws + off; off = (off + bytes + 255) & ~(size_t)255; return p; };
    const size_t act = (size_t)TN_ * C_ * HW_ * sizeof(unsigned short);  // 4 MB
    unsigned short* xT    = (unsigned short*)take(act);
    unsigned short* wb    = (unsigned short*)take((size_t)4 * C_ * C_ * sizeof(unsigned short));
    unsigned short* qb    = (unsigned short*)take(act);
    unsigned short* kb    = (unsigned short*)take(act);
    unsigned short* vb    = (unsigned short*)take(act);
    unsigned short* attnT = (unsigned short*)take(act);
    float* proj = (float*)take((size_t)TN_ * C_ * HW_ * sizeof(float));  // 8 MB
    float* sum  = (float*)take(C_ * sizeof(float));
    float* sqs  = (float*)take(C_ * sizeof(float));
    (void)in_sizes; (void)n_in; (void)out_size; (void)ws_size;

    {   // K0: converts + BN stat init
        int elems = TN_ * C_ * HW_;
        k_cvt_x<<<(elems + 255) / 256, 256, 0, stream>>>(x, xT);
        k_cvt_w<<<(4 * C_ * C_) / 256, 256, 0, stream>>>(Wq, Wk, Wv, Wo, wb);
        k_zero_stats<<<1, 256, 0, stream>>>(sum, sqs);
    }
    // K1: q,k,v projections
    k_qkv<<<dim3(HW_ / 16, TN_, 3), 128, 0, stream>>>(
        xT, wb, wb + C_ * C_, wb + 2 * C_ * C_, qb, kb, vb);
    // K2: fused attention + spiking LIF (scores stay on-chip)
    k_attn<<<dim3(HW_ / 64, NB_ * NH_), 128, 0, stream>>>(qb, kb, vb, attnT);
    // K3: output projection + BN statistics
    k_proj<<<dim3(HW_ / 16, TN_), 128, 0, stream>>>(attnT, wb + 3 * C_ * C_, proj, sum, sqs);
    // K4: BN + LIF + residual
    k_bnlif<<<(NB_ * C_ * HW_ + 255) / 256, 256, 0, stream>>>(
        proj, x, gamma, beta, sum, sqs, out);
}
